// StreamingTransformerLayer_3796751090427
// MI455X (gfx1250) — matrix operand order repaired and checked by reading, not compiled, not run on hardware
//
#include <hip/hip_runtime.h>
#include <hip/hip_bf16.h>

// ---------------------------------------------------------------------------
// CDNA5 (gfx1250) wave32 WMMA transformer layer.
// Dense math via v_wmma_f32_16x16x32_bf16; B tiles staged through LDS with
// async global->LDS copies, two-stage ping-pong double buffering.
// ---------------------------------------------------------------------------

typedef __attribute__((ext_vector_type(16))) __bf16 v16bf;
typedef __attribute__((ext_vector_type(8)))  float  v8f;

union BF16x16 {
    v16bf v;
    uint4 u[2];
};

__device__ __forceinline__ v8f wmma_bf16(const v16bf& a, const v16bf& b, v8f c) {
    return __builtin_amdgcn_wmma_f32_16x16x32_bf16(false, a, false, b, (short)0, c, false, false);
}

#if defined(__HIP_DEVICE_COMPILE__) && __has_builtin(__builtin_amdgcn_global_load_async_to_lds_b128)
#define USE_ASYNC_LDS 1
#else
#define USE_ASYNC_LDS 0
#endif

#if USE_ASYNC_LDS
// Parameter types per the compiler diagnostic: AS1 / AS3 pointers to int4.
typedef int v4i_vs __attribute__((vector_size(16)));
typedef __attribute__((address_space(1))) v4i_vs* as1_v4i;
typedef __attribute__((address_space(3))) v4i_vs* as3_v4i;

// Copy 32 bf16 (64 bytes) global -> LDS, asynchronously (4 x b128).
__device__ __forceinline__ void async_copy_64B(const __bf16* gsrc, __bf16* ldst) {
    #pragma unroll
    for (int j = 0; j < 4; ++j) {
        __builtin_amdgcn_global_load_async_to_lds_b128(
            (as1_v4i)(gsrc + 8 * j), (as3_v4i)(ldst + 8 * j), 0, 0);
    }
}

template <int Ncnt>
__device__ __forceinline__ void wait_asynccnt() {
#if __has_builtin(__builtin_amdgcn_s_wait_asynccnt)
    __builtin_amdgcn_s_wait_asynccnt(Ncnt);
#else
    if constexpr (Ncnt == 0) asm volatile("s_wait_asynccnt 0x0" ::: "memory");
    else                     asm volatile("s_wait_asynccnt 0x4" ::: "memory");
#endif
}
#endif

// ---------------------------------------------------------------------------
// Weight convert + transpose: W [N][K] f32  ->  Wt [K][N] bf16
// ---------------------------------------------------------------------------
__global__ void wt_transpose_kernel(const float* __restrict__ W,
                                    __bf16* __restrict__ Wt,
                                    int N, int K) {
    size_t idx = (size_t)blockIdx.x * blockDim.x + threadIdx.x;
    size_t total = (size_t)N * K;
    if (idx >= total) return;
    int k = (int)(idx % K);
    int n = (int)(idx / K);
    Wt[(size_t)k * N + n] = (__bf16)W[idx];
}

// ---------------------------------------------------------------------------
// LayerNorm over E, output bf16. One 256-thread block per row.
// ---------------------------------------------------------------------------
__global__ __launch_bounds__(256)
void layernorm_kernel(const float* __restrict__ X,
                      const float* __restrict__ g,
                      const float* __restrict__ be,
                      __bf16* __restrict__ out, int E) {
    __shared__ float red[256];
    const int row = blockIdx.x;
    const int tid = threadIdx.x;
    const float* x = X + (size_t)row * E;

    float s = 0.f;
    for (int j = tid; j < E; j += 256) s += x[j];
    red[tid] = s; __syncthreads();
    for (int off = 128; off > 0; off >>= 1) {
        if (tid < off) red[tid] += red[tid + off];
        __syncthreads();
    }
    const float mu = red[0] / (float)E;
    __syncthreads();

    float v = 0.f;
    for (int j = tid; j < E; j += 256) { float d = x[j] - mu; v += d * d; }
    red[tid] = v; __syncthreads();
    for (int off = 128; off > 0; off >>= 1) {
        if (tid < off) red[tid] += red[tid + off];
        __syncthreads();
    }
    const float var = red[0] / (float)E;
    const float rs = rsqrtf(var + 1e-5f);

    __bf16* o = out + (size_t)row * E;
    for (int j = tid; j < E; j += 256)
        o[j] = (__bf16)((x[j] - mu) * rs * g[j] + be[j]);
}

// ---------------------------------------------------------------------------
// WMMA GEMM: C[M][N] = A[M][K](bf16) @ Bt[K][N](bf16)
// Block = 4 waves, tile 128(M) x 128(N); wave = 32(M) x 128(N) = 16 WMMA/step.
// B tile (32K x 128N) staged in LDS, shared by all waves, explicit two-stage
// ping-pong double buffering (K % 64 == 0 required). B fragments from LDS are
// software-pipelined (preload nt+1 while computing nt).
// mode: 0 = f32 out, 1 = f32 + residual, 2 = bf16 out with exact GELU.
// ---------------------------------------------------------------------------
constexpr int GPITCH = 136;  // 128 + 8 bf16 pad (16B aligned rows, spreads banks)

__device__ __forceinline__ void gemm_load_a(const uint4* arow0, const uint4* arow1,
                                            int k0, int hi,
                                            BF16x16& a0, BF16x16& a1) {
    const uint4* p0 = arow0 + (k0 >> 3);
    a0.u[0] = p0[hi]; a0.u[1] = p0[2 + hi];
    const uint4* p1 = arow1 + (k0 >> 3);
    a1.u[0] = p1[hi]; a1.u[1] = p1[2 + hi];
}

__device__ __forceinline__ void gemm_compute_tile(const __bf16* sbuf, int lane,
                                                  const BF16x16& a0, const BF16x16& a1,
                                                  v8f (&acc)[2][8]) {
    const int bhi = lane >> 4;
    const __bf16* bbase = sbuf + (lane & 15);
    BF16x16 bf[2];
    {
        #pragma unroll
        for (int s = 0; s < 16; ++s)
            bf[0].v[s] = bbase[((s & 7) + 8 * bhi + 16 * (s >> 3)) * GPITCH];
    }
    #pragma unroll
    for (int nt = 0; nt < 8; ++nt) {
        if (nt < 7) {
            #pragma unroll
            for (int s = 0; s < 16; ++s)
                bf[(nt + 1) & 1].v[s] = bbase[((s & 7) + 8 * bhi + 16 * (s >> 3)) * GPITCH + (nt + 1) * 16];
        }
        acc[0][nt] = wmma_bf16(a0.v, bf[nt & 1].v, acc[0][nt]);
        acc[1][nt] = wmma_bf16(a1.v, bf[nt & 1].v, acc[1][nt]);
    }
}

__global__ __launch_bounds__(128)
void gemm_wmma_kernel(const __bf16* __restrict__ A,
                      const __bf16* __restrict__ Bt,
                      float* __restrict__ outF,
                      __bf16* __restrict__ outB,
                      const float* __restrict__ res,
                      int M, int N, int K, int mode) {
    __shared__ __align__(16) __bf16 sB[2][32 * GPITCH];

    const int tid  = threadIdx.x;
    const int wave = tid >> 5;
    const int lane = tid & 31;
    const int hi = lane >> 4;
    const int lo = lane & 15;
    const int m0 = blockIdx.x * 128 + wave * 32;
    const int n0 = blockIdx.y * 128;

    // B-tile cooperative copy: 128 threads x 64B = 32(K) x 128(N) bf16
    const int rB = tid >> 2;          // tile row 0..31 (K)
    const int cB = (tid & 3) * 32;    // col chunk 0/32/64/96
    const __bf16* bsrc0 = Bt + (size_t)rB * N + n0 + cB;
    __bf16* bdst0 = &sB[0][rB * GPITCH + cB];
    __bf16* bdst1 = &sB[1][rB * GPITCH + cB];

    v8f acc[2][8] = {};

    const uint4* arow0 = (const uint4*)(A + (size_t)(m0 + lo) * K);
    const uint4* arow1 = (const uint4*)(A + (size_t)(m0 + 16 + lo) * K);

    BF16x16 aS0_0, aS0_1, aS1_0, aS1_1;

#if !USE_ASYNC_LDS
    uint4 breg[4];
    auto load_breg = [&](const __bf16* src) {
        const uint4* gp = (const uint4*)src;
        breg[0] = gp[0]; breg[1] = gp[1]; breg[2] = gp[2]; breg[3] = gp[3];
    };
    auto store_breg = [&](__bf16* dst) {
        uint4* lp = (uint4*)dst;
        lp[0] = breg[0]; lp[1] = breg[1]; lp[2] = breg[2]; lp[3] = breg[3];
    };
#endif

    // Prologue: stage k = 0 into buf0
#if USE_ASYNC_LDS
    async_copy_64B(bsrc0, bdst0);
#else
    load_breg(bsrc0);
    store_breg(bdst0);
#endif
    gemm_load_a(arow0, arow1, 0, hi, aS0_0, aS0_1);

    // K % 64 == 0: explicit two-stage ping-pong, no register copies.
    for (int k0 = 0; k0 < K; k0 += 64) {
        // ---- stage 0: compute k0 from buf0, prefetch k0+32 into buf1 ----
        {
#if USE_ASYNC_LDS
            async_copy_64B(bsrc0 + (size_t)(k0 + 32) * N, bdst1);
#else
            load_breg(bsrc0 + (size_t)(k0 + 32) * N);
#endif
            gemm_load_a(arow0, arow1, k0 + 32, hi, aS1_0, aS1_1);
#if USE_ASYNC_LDS
            wait_asynccnt<4>();
#endif
            __syncthreads();
            gemm_compute_tile(sB[0], lane, aS0_0, aS0_1, acc);
            __syncthreads();
#if !USE_ASYNC_LDS
            store_breg(bdst1);
#endif
        }
        // ---- stage 1: compute k0+32 from buf1, prefetch k0+64 into buf0 ----
        {
            const bool hasNext = (k0 + 64) < K;
            if (hasNext) {
#if USE_ASYNC_LDS
                async_copy_64B(bsrc0 + (size_t)(k0 + 64) * N, bdst0);
#else
                load_breg(bsrc0 + (size_t)(k0 + 64) * N);
#endif
                gemm_load_a(arow0, arow1, k0 + 64, hi, aS0_0, aS0_1);
            }
#if USE_ASYNC_LDS
            if (hasNext) wait_asynccnt<4>(); else wait_asynccnt<0>();
#endif
            __syncthreads();
            gemm_compute_tile(sB[1], lane, aS1_0, aS1_1, acc);
            __syncthreads();
#if !USE_ASYNC_LDS
            if (hasNext) store_breg(bdst0);
#endif
        }
    }

    #pragma unroll
    for (int mt = 0; mt < 2; ++mt) {
        #pragma unroll
        for (int nt = 0; nt < 8; ++nt) {
            #pragma unroll
            for (int r = 0; r < 8; ++r) {
                const int row = m0 + mt * 16 + r + 8 * hi;
                const int col = n0 + nt * 16 + lo;
                const size_t idx = (size_t)row * N + col;
                const float val = acc[mt][nt][r];
                if (mode == 0) {
                    outF[idx] = val;
                } else if (mode == 1) {
                    outF[idx] = val + res[idx];
                } else {
                    const float ge = 0.5f * val * (1.0f + erff(val * 0.70710678118654752f));
                    outB[idx] = (__bf16)ge;
                }
            }
        }
    }
}

// ---------------------------------------------------------------------------
// RoPE + QKV split. proj f32 [B,T,3,H,D] ->
//   Q  bf16 [B,H,T,D] (rotated), KT bf16 [B,H,D,T] (rotated, transposed),
//   V  bf16 [B,H,T,D]
// ---------------------------------------------------------------------------
__global__ void rope_split_kernel(const float* __restrict__ proj,
                                  __bf16* __restrict__ Q,
                                  __bf16* __restrict__ KT,
                                  __bf16* __restrict__ V,
                                  int Bb, int T, int H) {
    const int D = 64, half = 32;
    size_t idx = (size_t)blockIdx.x * blockDim.x + threadIdx.x;
    const size_t total = (size_t)Bb * T * H * half;
    if (idx >= total) return;

    const int i = (int)(idx % half);
    size_t r = idx / half;
    const int h = (int)(r % H); r /= H;
    const int t = (int)(r % T);
    const int b = (int)(r / T);

    const size_t qbase = ((((size_t)b * T + t) * 3 + 0) * H + h) * D;
    const size_t kbase = ((((size_t)b * T + t) * 3 + 1) * H + h) * D;
    const size_t vbase = ((((size_t)b * T + t) * 3 + 2) * H + h) * D;

    const float qr = proj[qbase + 2 * i], qi = proj[qbase + 2 * i + 1];
    const float kr = proj[kbase + 2 * i], ki = proj[kbase + 2 * i + 1];
    const float vr = proj[vbase + 2 * i], vi = proj[vbase + 2 * i + 1];

    const float freq = __powf(10000.0f, -(float)i / (float)half);
    const float ang = (float)t * freq;
    float sn, cs;
    __sincosf(ang, &sn, &cs);

    const size_t bh = (size_t)b * H + h;
    __bf16* qd = Q + (bh * T + t) * D;
    qd[2 * i]     = (__bf16)(qr * cs - qi * sn);
    qd[2 * i + 1] = (__bf16)(qr * sn + qi * cs);

    __bf16* vd = V + (bh * T + t) * D;
    vd[2 * i]     = (__bf16)vr;
    vd[2 * i + 1] = (__bf16)vi;

    __bf16* kd = KT + bh * (size_t)D * T;
    kd[(size_t)(2 * i) * T + t]     = (__bf16)(kr * cs - ki * sn);
    kd[(size_t)(2 * i + 1) * T + t] = (__bf16)(kr * sn + ki * cs);
}

// ---------------------------------------------------------------------------
// Flash-style sliding-window attention. One wave per (b*H, 16-query tile).
// ---------------------------------------------------------------------------
__global__ __launch_bounds__(32)
void attn_kernel(const __bf16* __restrict__ Q,
                 const __bf16* __restrict__ KT,
                 const __bf16* __restrict__ V,
                 __bf16* __restrict__ O,
                 int T, int H, int Wctx) {
    __shared__ __align__(16) __bf16 lds[16 * 32];

    const int D = 64;
    const float scale = 0.125f; // 1/sqrt(64)
    const int lane = threadIdx.x;
    const int hi = lane >> 4;
    const int lo = lane & 15;
    const int bh = blockIdx.x;
    const int tq0 = blockIdx.y * 16;

    const __bf16* qb  = Q  + (size_t)bh * T * D;
    const __bf16* ktb = KT + (size_t)bh * D * T;
    const __bf16* vb  = V  + (size_t)bh * T * D;

    BF16x16 aq0, aq1;
    {
        const uint4* qp = (const uint4*)(qb + (size_t)(tq0 + lo) * D);
        aq0.u[0] = qp[hi];     aq0.u[1] = qp[2 + hi];
        aq1.u[0] = qp[4 + hi]; aq1.u[1] = qp[6 + hi];
    }

    float mrow[8], lrow[8];
    #pragma unroll
    for (int r = 0; r < 8; ++r) { mrow[r] = -3.0e38f; lrow[r] = 0.f; }
    v8f accO[4] = {};

    int kb0 = tq0 - (Wctx - 1);
    if (kb0 < 0) kb0 = 0;
    kb0 &= ~31;

    for (int kb = kb0; kb <= tq0 + 15; kb += 32) {
        BF16x16 bk00, bk01, bk10, bk11;
        {
            #pragma unroll
            for (int s = 0; s < 16; ++s) {
                const int d = (s & 7) + 8 * hi + 16 * (s >> 3);
                const __bf16* p0 = ktb + (size_t)d * T + kb + lo;
                const __bf16* p1 = ktb + (size_t)(32 + d) * T + kb + lo;
                bk00.v[s] = p0[0]; bk01.v[s] = p0[16];
                bk10.v[s] = p1[0]; bk11.v[s] = p1[16];
            }
        }
        v8f sa = {}, sb = {};
        sa = wmma_bf16(aq0.v, bk00.v, sa);
        sa = wmma_bf16(aq1.v, bk10.v, sa);
        sb = wmma_bf16(aq0.v, bk01.v, sb);
        sb = wmma_bf16(aq1.v, bk11.v, sb);

        float fscale[8];
        #pragma unroll
        for (int r = 0; r < 8; ++r) {
            const int t = tq0 + r + 8 * hi;
            const int s0 = kb + lo;
            const int s1 = kb + 16 + lo;
            const bool v0 = (s0 <= t) && ((t - s0) < Wctx);
            const bool v1 = (s1 <= t) && ((t - s1) < Wctx);
            const float x0 = v0 ? sa[r] * scale : -3.0e38f;
            const float x1 = v1 ? sb[r] * scale : -3.0e38f;

            float mx = fmaxf(x0, x1);
            #pragma unroll
            for (int msk = 1; msk <= 8; msk <<= 1)
                mx = fmaxf(mx, __shfl_xor(mx, msk, 32));
            const float mnew = fmaxf(mrow[r], mx);

            const float f = __expf(mrow[r] - mnew);
            const float e0 = v0 ? __expf(x0 - mnew) : 0.f;
            const float e1 = v1 ? __expf(x1 - mnew) : 0.f;

            float rs = e0 + e1;
            #pragma unroll
            for (int msk = 1; msk <= 8; msk <<= 1)
                rs += __shfl_xor(rs, msk, 32);

            lrow[r] = lrow[r] * f + rs;
            mrow[r] = mnew;
            fscale[r] = f;

            lds[(r + 8 * hi) * 32 + lo]      = (__bf16)e0;
            lds[(r + 8 * hi) * 32 + 16 + lo] = (__bf16)e1;
        }
        __syncthreads();

        BF16x16 ap;
        {
            const uint4* lp = (const uint4*)(lds + lo * 32);
            ap.u[0] = lp[hi];
            ap.u[1] = lp[2 + hi];
        }

        #pragma unroll
        for (int c = 0; c < 4; ++c)
            #pragma unroll
            for (int r = 0; r < 8; ++r)
                accO[c][r] *= fscale[r];

        #pragma unroll
        for (int c = 0; c < 4; ++c) {
            BF16x16 bv;
            #pragma unroll
            for (int s = 0; s < 16; ++s)
                bv.v[s] = vb[(size_t)(kb + (s & 7) + 8 * hi + 16 * (s >> 3)) * D + c * 16 + lo];
            accO[c] = wmma_bf16(ap.v, bv.v, accO[c]);
        }
        __syncthreads();
    }

    float inv[8];
    #pragma unroll
    for (int r = 0; r < 8; ++r) inv[r] = 1.0f / lrow[r];

    const int b = bh / H;
    const int h = bh % H;
    #pragma unroll
    for (int c = 0; c < 4; ++c) {
        #pragma unroll
        for (int r = 0; r < 8; ++r) {
            const int t = tq0 + r + 8 * hi;
            const int d = c * 16 + lo;
            O[(((size_t)b * T + t) * H + h) * (size_t)D + d] = (__bf16)(accO[c][r] * inv[r]);
        }
    }
}

// ---------------------------------------------------------------------------
// Launcher
// ---------------------------------------------------------------------------
extern "C" void kernel_launch(void* const* d_in, const int* in_sizes, int n_in,
                              void* d_out, int out_size, void* d_ws, size_t ws_size,
                              hipStream_t stream) {
    (void)n_in; (void)out_size; (void)ws_size;

    const float* x      = (const float*)d_in[0];
    const float* w_qkv  = (const float*)d_in[1]; // (3E, E)
    const float* w_out  = (const float*)d_in[2]; // (E, E)
    const float* ln1_g  = (const float*)d_in[3];
    const float* ln1_b  = (const float*)d_in[4];
    const float* ln2_g  = (const float*)d_in[5];
    const float* ln2_b  = (const float*)d_in[6];
    const float* w1     = (const float*)d_in[7]; // (F, E)
    const float* w2     = (const float*)d_in[8]; // (E, F)

    const int E = 1024, H = 16, F = 4096, T = 2048;
    const int N3 = 3 * E;
    const int M  = in_sizes[0] / E;   // B*T = 4096
    const int Bb = M / T;             // 2
    const int Wctx = 256;

    char* ws = (char*)d_ws;
    size_t off = 0;
    auto alloc = [&](size_t bytes) -> void* {
        void* p = ws + off;
        off = (off + bytes + 255) & ~(size_t)255;
        return p;
    };

    __bf16* WqkvT = (__bf16*)alloc((size_t)E * N3 * 2);
    __bf16* WoutT = (__bf16*)alloc((size_t)E * E * 2);
    __bf16* W1T   = (__bf16*)alloc((size_t)E * F * 2);
    __bf16* W2T   = (__bf16*)alloc((size_t)F * E * 2);
    __bf16* hbuf  = (__bf16*)alloc((size_t)M * E * 2);   // h1 / h2 (reused)
    __bf16* qbuf  = (__bf16*)alloc((size_t)M * E * 2);
    __bf16* ktbuf = (__bf16*)alloc((size_t)M * E * 2);
    __bf16* vbuf  = (__bf16*)alloc((size_t)M * E * 2);
    __bf16* obuf  = (__bf16*)alloc((size_t)M * E * 2);
    float*  x1    = (float*) alloc((size_t)M * E * 4);
    void*   pm    = alloc((size_t)M * N3 * 4);           // proj f32 | mid bf16 (aliased)
    float*  proj  = (float*)pm;
    __bf16* mid   = (__bf16*)pm;

    float* out = (float*)d_out;

    // 1. Weight convert + transpose
    {
        const int th = 256;
        size_t n;
        n = (size_t)N3 * E; wt_transpose_kernel<<<dim3((n + th - 1) / th), th, 0, stream>>>(w_qkv, WqkvT, N3, E);
        n = (size_t)E * E;  wt_transpose_kernel<<<dim3((n + th - 1) / th), th, 0, stream>>>(w_out, WoutT, E, E);
        n = (size_t)F * E;  wt_transpose_kernel<<<dim3((n + th - 1) / th), th, 0, stream>>>(w1, W1T, F, E);
        n = (size_t)E * F;  wt_transpose_kernel<<<dim3((n + th - 1) / th), th, 0, stream>>>(w2, W2T, E, F);
    }

    // 2. LN1
    layernorm_kernel<<<dim3(M), 256, 0, stream>>>(x, ln1_g, ln1_b, hbuf, E);

    // 3. QKV projection: proj[M][3E] = h @ WqkvT
    gemm_wmma_kernel<<<dim3(M / 128, N3 / 128), 128, 0, stream>>>(
        hbuf, WqkvT, proj, nullptr, nullptr, M, N3, E, 0);

    // 4. RoPE + split
    {
        const size_t total = (size_t)Bb * T * H * 32;
        rope_split_kernel<<<dim3((total + 255) / 256), 256, 0, stream>>>(
            proj, qbuf, ktbuf, vbuf, Bb, T, H);
    }

    // 5. Sliding-window flash attention -> O [B,T,H,D] == [M][E]
    attn_kernel<<<dim3(Bb * H, T / 16), 32, 0, stream>>>(
        qbuf, ktbuf, vbuf, obuf, T, H, Wctx);

    // 6. Out-projection + residual: x1 = x + O @ WoutT
    gemm_wmma_kernel<<<dim3(M / 128, E / 128), 128, 0, stream>>>(
        obuf, WoutT, x1, nullptr, x, M, E, E, 1);

    // 7. LN2
    layernorm_kernel<<<dim3(M), 256, 0, stream>>>(x1, ln2_g, ln2_b, hbuf, E);

    // 8. FFN up + exact GELU: mid[M][F] = gelu(h2 @ W1T)
    gemm_wmma_kernel<<<dim3(M / 128, F / 128), 128, 0, stream>>>(
        hbuf, W1T, nullptr, mid, nullptr, M, F, E, 2);

    // 9. FFN down + residual: out = x1 + mid @ W2T
    gemm_wmma_kernel<<<dim3(M / 128, E / 128), 128, 0, stream>>>(
        mid, W2T, out, nullptr, x1, M, E, F, 1);
}
